// _MLX4Linear_7009386627393
// MI455X (gfx1250) — matrix operand order repaired and checked by reading, not compiled, not run on hardware
//
#include <hip/hip_runtime.h>

// W4 grouped-quant GEMM for MI455X (gfx1250, wave32).
// out[32,14336] = x[32,4096] @ dequant(weight_q)^T
// Memory-bound: 117MB packed-weight stream / 23.3 TB/s ~= 5.3us floor.
// Strategy: single pass over weights, magic-bias fp16 dequant (pure pk ops),
// v_wmma_f32_16x16x32_f16 accumulate, split-K=2 for 224 blocks / 4 waves/SIMD.

typedef __attribute__((ext_vector_type(16))) _Float16 v16h;
typedef __attribute__((ext_vector_type(8)))  _Float16 v8h;
typedef __attribute__((ext_vector_type(4)))  _Float16 v4h;
typedef __attribute__((ext_vector_type(2)))  _Float16 v2h;
typedef __attribute__((ext_vector_type(8)))  float    v8f;

#define O_FEAT   14336
#define I_FEAT   4096
#define NTOK     32
#define GRP      64
#define KSPLIT   2
#define KHALF    (I_FEAT / KSPLIT)   // 2048 K per block
#define KTILES   (KHALF / 32)        // 64 K-tiles of 32 per block
#define WQ_COLS  (I_FEAT / 2)        // 2048 int32 per weight row (1 byte each)
#define NGROUPS  (I_FEAT / GRP)      // 64 scale groups per row
#define ROWH     (KHALF + 8)         // 2056 halves/row -> stride 1028 dwords = 4 mod 64 banks
#define LDS_BYTES (NTOK * ROWH * 2)  // 131584 B -> 2 blocks per 320KB WGP

struct WBuf { int4 w0, w1; _Float16 s, z; };

__device__ __forceinline__ WBuf load_w(const int* __restrict__ wrow,
                                       const _Float16* __restrict__ srow,
                                       const _Float16* __restrict__ zrow,
                                       int t, int grp) {
  WBuf b;
  const int4* p = (const int4*)(wrow + t * 16);   // wrow already includes +8*grp
  b.w0 = p[0];                                    // int32 cols c0..c0+3  -> K k0..k0+7
  b.w1 = p[1];                                    // cols c0+4..c0+7      -> K k0+8..k0+15
  const int g = (t * 32 + 16 * grp) >> 6;         // group within this K-half
  b.s = srow[g];
  b.z = zrow[g];
  return b;
}

// Magic-bias dequant: nibble q -> fp16 via (0x6400|q) = 1024+q (exact),
// subtract 1024 (exact), then packed fma.  Pure v_pk_* / bit ops, no f32 cvt.
__device__ __forceinline__ v16h dequant(const WBuf& wb) {
  v16h b;
  const v2h s2 = { wb.s, wb.s };
  const v2h z2 = { wb.z, wb.z };
  const v2h cneg = { (_Float16)(-1024.0f), (_Float16)(-1024.0f) };
  int w[8] = { wb.w0.x, wb.w0.y, wb.w0.z, wb.w0.w,
               wb.w1.x, wb.w1.y, wb.w1.z, wb.w1.w };
#pragma unroll
  for (int j = 0; j < 8; ++j) {
    unsigned u = (unsigned)w[j];                   // one byte: lo nibble=K2j, hi=K2j+1
    unsigned p = (((u << 12) | u) & 0x000F000Fu) | 0x64006400u;
    v2h q = __builtin_bit_cast(v2h, p);            // {1024+lo, 1024+hi}
    v2h r = (q + cneg) * s2 + z2;                  // v_pk_add_f16 + v_pk_fma_f16
    b[2 * j]     = r[0];
    b[2 * j + 1] = r[1];
  }
  return b;
}

// A-fragment (16x32 f16) from padded LDS rows.
__device__ __forceinline__ v16h load_a(const _Float16* p) {
  v8h lo = *(const v8h*)p;            // ds_load_b128
  v8h hi = *(const v8h*)(p + 8);      // ds_load_b128
  v16h a;
#pragma unroll
  for (int i = 0; i < 8; ++i) { a[i] = lo[i]; a[i + 8] = hi[i]; }
  return a;
}

__global__ void __launch_bounds__(256)
w4_gemm_wmma(const float* __restrict__ x,
             const int* __restrict__ wq,
             const _Float16* __restrict__ scales,
             const _Float16* __restrict__ zeros,
             float* __restrict__ out) {
  extern __shared__ _Float16 xs[];   // [32][2056] fp16: this block's K-half of x
  const int tid = threadIdx.x;
  const int kh  = blockIdx.y;        // K-half index

  // ---- Stage this K-half of x (fp32 -> fp16) into LDS once ----
  {
    const float4* x4 = (const float4*)x;
    for (int i = 0; i < (NTOK * KHALF / 4) / 256; ++i) {   // 64 iters
      const int idx = i * 256 + tid;
      const int row = idx >> 9;                  // / (KHALF/4)
      const int c4  = idx & 511;
      const float4 v = x4[row * (I_FEAT / 4) + kh * (KHALF / 4) + c4];
      v4h h = { (_Float16)v.x, (_Float16)v.y, (_Float16)v.z, (_Float16)v.w };
      *(v4h*)(xs + row * ROWH + c4 * 4) = h;
    }
  }
  __syncthreads();

  const int lane = tid & 31;
  const int wave = tid >> 5;
  const int grp  = lane >> 4;     // K half-group for A/B fragments
  const int n    = lane & 15;     // output column within N-tile / token row for A
  const int orow = blockIdx.x * 128 + wave * 16 + n;

  const int*      wrow = wq     + (size_t)orow * WQ_COLS + kh * (KHALF / 2) + 8 * grp;
  const _Float16* srow = scales + (size_t)orow * NGROUPS + kh * (KHALF / GRP);
  const _Float16* zrow = zeros  + (size_t)orow * NGROUPS + kh * (KHALF / GRP);
  const _Float16* abase = xs + n * ROWH + 16 * grp;

  v8f acc0 = {};   // tokens 0..15
  v8f acc1 = {};   // tokens 16..31

  WBuf cur = load_w(wrow, srow, zrow, 0, grp);

#define STEP(T, WB)                                                              \
  do {                                                                           \
    const _Float16* ap = abase + (T) * 32;                                       \
    v16h a0 = load_a(ap);                                                        \
    v16h a1 = load_a(ap + 16 * ROWH);                                            \
    v16h bf = dequant(WB);                                                       \
    acc0 = __builtin_amdgcn_wmma_f32_16x16x32_f16(false, a0, false, bf,          \
                                                  (short)0, acc0, false, false); \
    acc1 = __builtin_amdgcn_wmma_f32_16x16x32_f16(false, a1, false, bf,          \
                                                  (short)0, acc1, false, false); \
  } while (0)

  for (int t = 0; t < KTILES - 1; ++t) {
    WBuf nxt = load_w(wrow, srow, zrow, t + 1, grp);   // issue next tile's loads early
    const int tp = (t + 8 < KTILES) ? t + 8 : KTILES - 1;
    __builtin_prefetch(wrow + tp * 16, 0, 0);          // global_prefetch_b8
    STEP(t, cur);
    cur = nxt;
  }
  STEP(KTILES - 1, cur);
#undef STEP

  // ---- Epilogue: C/D layout M = r + 8*grp (+16 for acc1), N = lane&15.
  // Two K-half blocks contribute -> hardware f32 atomic add (out zeroed per call).
  float* o0 = out + (size_t)(8 * grp) * O_FEAT + orow;
#pragma unroll
  for (int r = 0; r < 8; ++r) {
    unsafeAtomicAdd(&o0[(size_t)r * O_FEAT],        acc0[r]);
    unsafeAtomicAdd(&o0[(size_t)(r + 16) * O_FEAT], acc1[r]);
  }
}

extern "C" void kernel_launch(void* const* d_in, const int* in_sizes, int n_in,
                              void* d_out, int out_size, void* d_ws, size_t ws_size,
                              hipStream_t stream) {
  const float*    x  = (const float*)d_in[0];
  const int*      wq = (const int*)d_in[1];
  const _Float16* sc = (const _Float16*)d_in[2];
  const _Float16* zr = (const _Float16*)d_in[3];
  float*          out = (float*)d_out;

  (void)in_sizes; (void)n_in; (void)d_ws; (void)ws_size;

  hipFuncSetAttribute((const void*)w4_gemm_wmma,
                      hipFuncAttributeMaxDynamicSharedMemorySize, LDS_BYTES);

  hipMemsetAsync(out, 0, (size_t)out_size * sizeof(float), stream);

  dim3 grid(O_FEAT / 128, KSPLIT);   // 224 blocks: 112 O-tiles x 2 K-halves
  dim3 block(256);                   // 8 waves; wave -> one 16-wide N-tile
  w4_gemm_wmma<<<grid, block, LDS_BYTES, stream>>>(x, wq, sc, zr, out);
}
